// GeometricAttentionLayer_55130200212213
// MI455X (gfx1250) — matrix operand order repaired and checked by reading, not compiled, not run on hardware
//
#include <hip/hip_runtime.h>
#include <hip/hip_bf16.h>
#include <math.h>

// ---------- types ----------
typedef __attribute__((ext_vector_type(16))) __bf16        v16bf;
typedef __attribute__((ext_vector_type(8)))  float         v8f;
typedef __attribute__((ext_vector_type(4)))  unsigned int  v4u;
typedef __attribute__((ext_vector_type(4)))  int           v4i;
typedef __attribute__((ext_vector_type(8)))  int           v8i;

__device__ __forceinline__ unsigned short f32_to_bf16(float f) {
  unsigned int u = __float_as_uint(f);
  u += 0x7fffu + ((u >> 16) & 1u);          // round-to-nearest-even
  return (unsigned short)(u >> 16);
}
__device__ __forceinline__ float bf16_to_f32(unsigned short h) {
  return __uint_as_float(((unsigned int)h) << 16);
}

__constant__ int c_ip[8] = {0, 2, 3, 4, 8, 9, 10, 14};

// ---------- TDM: 2D tile load global -> LDS (rows x tile_dwords, 4-byte units) ----------
// Descriptor per CDNA5 ISA 8.3/8.4: group0 = {count|flags, lds_addr, global_addr, type=2},
// group1 = {data_size, tensor_dim0/1, tile_dim0/1, tensor_dim0_stride}.
__device__ __forceinline__ void tdm_load_2d(unsigned lds_addr, const void* gptr,
                                            unsigned tile_rows, unsigned tile_dwords,
                                            unsigned row_stride_dwords) {
  unsigned long long ga = (unsigned long long)gptr;
  v4u g0;
  g0.x = 1u;                                             // count=1, no gather/iterate
  g0.y = lds_addr;                                       // LDS byte address
  g0.z = (unsigned)(ga & 0xffffffffu);                   // global_addr[31:0]
  g0.w = (unsigned)((ga >> 32) & 0x1ffffffu) | (2u << 30); // global_addr[56:32] | type=2
  unsigned td0 = row_stride_dwords;                      // tensor_dim0 >= tile_dim0
  unsigned td1 = tile_rows;                              // tensor_dim1
  v8i g1;
  g1[0] = (int)(2u << 16);                               // data_size = 2 (4 bytes)
  g1[1] = (int)((td0 & 0xffffu) << 16);                  // tensor_dim0[15:0]
  g1[2] = (int)((td0 >> 16) | ((td1 & 0xffffu) << 16));  // tensor_dim0[31:16] | tensor_dim1[15:0]
  g1[3] = (int)((td1 >> 16) | (tile_dwords << 16));      // tensor_dim1[31:16] | tile_dim0
  g1[4] = (int)(tile_rows & 0xffffu);                    // tile_dim1 (tile_dim2 = 0)
  g1[5] = (int)row_stride_dwords;                        // tensor_dim0_stride[31:0]
  g1[6] = 0;                                             // stride0[47:32] | stride1[15:0]
  g1[7] = 0;
  v4i z4 = {0, 0, 0, 0};
#if defined(__clang_major__) && (__clang_major__ >= 23)
  v8i z8 = {0, 0, 0, 0, 0, 0, 0, 0};
  __builtin_amdgcn_tensor_load_to_lds(g0, g1, z4, z4, z8, 0);
#else
  __builtin_amdgcn_tensor_load_to_lds(g0, g1, z4, z4, 0);
#endif
}

// ---------- 1) fuse blade into weights, TRANSPOSED:  Wt[n=j*16+y][k=i*16+x] ----------
__global__ __launch_bounds__(256) void fuse_weights_kernel(
    const float* __restrict__ w, const float* __restrict__ blade,
    unsigned short* __restrict__ Wt, int J, int I, long total) {
  long idx = (long)blockIdx.x * 256 + threadIdx.x;
  if (idx >= total) return;
  long Kdim = (long)I * 16;
  int k = (int)(idx % Kdim);
  long n = idx / Kdim;
  int j = (int)(n >> 4), y = (int)(n & 15);
  int i = k >> 4, xc = k & 15;
  float s = 0.f;
#pragma unroll
  for (int b = 0; b < 9; ++b)
    s += w[((long)j * I + i) * 9 + b] * blade[(b * 16 + xc) * 16 + y];
  Wt[idx] = f32_to_bf16(s);
}

// ---------- 2) fp32 -> bf16 elementwise ----------
__global__ __launch_bounds__(256) void cvt_f32_bf16_kernel(
    const float* __restrict__ x, unsigned short* __restrict__ y, long n) {
  long i = (long)blockIdx.x * 256 + threadIdx.x;
  if (i < n) y[i] = f32_to_bf16(x[i]);
}

// ---------- 3) bf16 GEMM, fp32 accumulate, WMMA 16x16x32, TDM-staged, double-buffered ----------
// C[z] = alpha * A[z] * Bt[z]^T ; A:[M,K] row-major bf16, Bt:[N,K] row-major bf16 (K-major B)
// Block tile 128x128, BK=64. 8 waves; wave = (rg 0..3, cg 0..1) computes 32x64 (8 WMMA tiles).
__global__ __launch_bounds__(256) void gemm_bf16_wmma_kernel(
    const unsigned short* __restrict__ A, const unsigned short* __restrict__ Bt,
    float* __restrict__ Cf, unsigned short* __restrict__ Cb,
    int M, int N, int K,
    long sA, int dA, long sB, int dB, long sC, float alpha) {
  int z = blockIdx.z;
  A  += (long)(z / dA) * sA;
  Bt += (long)(z / dB) * sB;
  long coff = (long)z * sC;

  __shared__ __align__(16) unsigned short lA[2][128 * 64];   // [m][k]
  __shared__ __align__(16) unsigned short lB[2][128 * 64];   // [n][k]

  const int t      = threadIdx.x;
  const int lane   = t & 31;
  const int laneHi = lane >> 4;   // 0/1
  const int lmod   = lane & 15;
  const int wave   = t >> 5;      // 0..7
  const int rg     = wave & 3;    // 32-row group
  const int cg     = wave >> 2;   // 64-col group

  const int m0 = blockIdx.y * 128;
  const int n0 = blockIdx.x * 128;
  const unsigned strideDw = (unsigned)(K >> 1);   // row stride in dwords

  auto issue = [&](int buf, int k0) {
    tdm_load_2d((unsigned)(size_t)&lA[buf][0], A + (size_t)m0 * K + k0, 128, 32, strideDw);
    tdm_load_2d((unsigned)(size_t)&lB[buf][0], Bt + (size_t)n0 * K + k0, 128, 32, strideDw);
  };

  v8f acc[2][4] = {};

  if (wave == 0) issue(0, 0);
  __builtin_amdgcn_s_wait_tensorcnt(0);
  __syncthreads();

  const int R = K >> 6;            // k-rounds of 64
  for (int r = 0; r < R; ++r) {
    int cur = r & 1;
    if (wave == 0 && r + 1 < R) issue(1 - cur, (r + 1) << 6);

#pragma unroll
    for (int ks = 0; ks < 2; ++ks) {           // two 16x16x32 k-steps within BK=64
      union Frag { v16bf v; uint4 q[2]; };
      Frag a[2], b[4];
#pragma unroll
      for (int s = 0; s < 2; ++s) {
        int mrow = (rg * 32 + s * 16 + lmod) * 64 + ks * 32;
        a[s].q[0] = *(const uint4*)&lA[cur][mrow + laneHi * 8];        // K = hi*8 + 0..7 pairs
        a[s].q[1] = *(const uint4*)&lA[cur][mrow + 16 + laneHi * 8];   // K = 16 + ...
      }
#pragma unroll
      for (int s = 0; s < 4; ++s) {
        int nrow = (cg * 64 + s * 16 + lmod) * 64 + ks * 32;
        b[s].q[0] = *(const uint4*)&lB[cur][nrow + laneHi * 8];        // K = hi*8 + 0..7
        b[s].q[1] = *(const uint4*)&lB[cur][nrow + 16 + laneHi * 8];   // K = 16 + hi*8 + 0..7
      }
#pragma unroll
      for (int sa = 0; sa < 2; ++sa)
#pragma unroll
        for (int sb = 0; sb < 4; ++sb)
          acc[sa][sb] = __builtin_amdgcn_wmma_f32_16x16x32_bf16(
              false, a[sa].v, false, b[sb].v, (short)0, acc[sa][sb], false, false);
    }

    __builtin_amdgcn_s_wait_tensorcnt(0);   // next-round tiles landed (no-op for waves 1..7)
    __syncthreads();                        // everyone done reading cur before it is reused
  }

  // C/D layout: VGPR v -> M = v + 8*laneHi ; N = lane&15
#pragma unroll
  for (int sa = 0; sa < 2; ++sa)
#pragma unroll
    for (int sb = 0; sb < 4; ++sb)
#pragma unroll
      for (int v = 0; v < 8; ++v) {
        int m = m0 + rg * 32 + sa * 16 + v + laneHi * 8;
        int n = n0 + cg * 64 + sb * 16 + lmod;
        float val = alpha * acc[sa][sb][v];
        if (Cf) Cf[coff + (long)m * N + n] = val;
        else    Cb[coff + (long)m * N + n] = f32_to_bf16(val);
      }
}

// ---------- 4) gather qm[b,h,n, d*8+ii] = q[b,n, (d*8+h)*16 + IP[ii]] ----------
__global__ __launch_bounds__(256) void gather_qm_kernel(
    const unsigned short* __restrict__ q, unsigned short* __restrict__ qm, long total) {
  long idx = (long)blockIdx.x * 256 + threadIdx.x;
  if (idx >= total) return;
  int ii = (int)(idx & 7);
  int d  = (int)((idx >> 3) & 63);
  int n  = (int)((idx >> 9) & 2047);
  int h  = (int)((idx >> 20) & 7);
  int b  = (int)(idx >> 23);
  qm[idx] = q[((long)(b * 2048 + n)) * 8192 + (d * 8 + h) * 16 + c_ip[ii]];
}

// ---------- 5) gather km[b, nk, d*8+ii] = k[b,nk, d*16 + IP[ii]]  (Bt layout [key][feat]) ----------
__global__ __launch_bounds__(256) void gather_km_kernel(
    const unsigned short* __restrict__ k, unsigned short* __restrict__ km, long total) {
  long idx = (long)blockIdx.x * 256 + threadIdx.x;
  if (idx >= total) return;
  int f  = (int)(idx & 511);
  int nk = (int)((idx >> 9) & 2047);
  int b  = (int)(idx >> 20);
  int d = f >> 3, ii = f & 7;
  km[idx] = k[((long)(b * 2048 + nk)) * 1024 + d * 16 + c_ip[ii]];
}

// ---------- 6) vT[b, f, nk] = v[b, nk, f]  (Bt layout for P@V) ----------
__global__ __launch_bounds__(256) void transpose_v_kernel(
    const unsigned short* __restrict__ v, unsigned short* __restrict__ vT, long total) {
  long idx = (long)blockIdx.x * 256 + threadIdx.x;
  if (idx >= total) return;
  int nk = (int)(idx & 2047);
  int f  = (int)((idx >> 11) & 1023);
  int b  = (int)(idx >> 21);
  vT[idx] = v[((long)(b * 2048 + nk)) * 1024 + f];
}

// ---------- 7) row softmax in place on bf16 scores, L=2048, one block/row ----------
__global__ __launch_bounds__(256) void softmax_rows_kernel(unsigned short* __restrict__ S, int L) {
  long row = blockIdx.x;
  unsigned short* p = S + row * (long)L;
  __shared__ float red[256];
  int t = threadIdx.x;

  float m = -3.0e38f;
  for (int i = t; i < L; i += 256) m = fmaxf(m, bf16_to_f32(p[i]));
  red[t] = m; __syncthreads();
  for (int s = 128; s > 0; s >>= 1) { if (t < s) red[t] = fmaxf(red[t], red[t + s]); __syncthreads(); }
  m = red[0]; __syncthreads();

  float vals[8];
  float sum = 0.f;
  int c = 0;
  for (int i = t; i < L; i += 256, ++c) {
    float e = __expf(bf16_to_f32(p[i]) - m);
    vals[c] = e; sum += e;
  }
  red[t] = sum; __syncthreads();
  for (int s = 128; s > 0; s >>= 1) { if (t < s) red[t] += red[t + s]; __syncthreads(); }
  float inv = 1.f / red[0];
  c = 0;
  for (int i = t; i < L; i += 256, ++c) p[i] = f32_to_bf16(vals[c] * inv);
}

// ---------- 8) regather outflat[b*N+n, (h*64+d)*16+c] = ao[b,h,n, d*16+c] ----------
__global__ __launch_bounds__(256) void regather_out_kernel(
    const unsigned short* __restrict__ ao, unsigned short* __restrict__ of, long total) {
  long idx = (long)blockIdx.x * 256 + threadIdx.x;
  if (idx >= total) return;
  int f = (int)(idx & 8191);
  long row = idx >> 13;            // b*2048 + n
  int b = (int)(row >> 11), n = (int)(row & 2047);
  int cc = f & 15;
  int hd = f >> 4;
  int h = hd >> 6, d = hd & 63;
  of[idx] = ao[(((long)(b * 8 + h) * 2048 + n) * 1024) + d * 16 + cc];
}

// ================= host side =================
extern "C" void kernel_launch(void* const* d_in, const int* in_sizes, int n_in,
                              void* d_out, int out_size, void* d_ws, size_t ws_size,
                              hipStream_t stream) {
  (void)in_sizes; (void)n_in; (void)out_size; (void)ws_size;

  const float* x     = (const float*)d_in[0];   // [2,2048,64,16]
  const float* blade = (const float*)d_in[1];   // [9,16,16]
  const float* w_q   = (const float*)d_in[2];   // [512,64,9]
  const float* w_k   = (const float*)d_in[3];   // [64,64,9]
  const float* w_v   = (const float*)d_in[4];   // [64,64,9]
  const float* w_o   = (const float*)d_in[5];   // [64,512,9]
  float* out = (float*)d_out;                   // [2,2048,64,16] fp32

  const int B = 2, N = 2048, Hh = 8;
  const long M = (long)B * N;                   // 4096

  // ---- workspace layout (bf16 buffers) ----
  char* ws = (char*)d_ws;
  size_t off = 0;
  auto alloc = [&](size_t bytes) -> size_t {
    off = (off + 255) & ~(size_t)255;
    size_t o = off; off += bytes; return o;
  };
  size_t oWq  = alloc((size_t)8192 * 1024 * 2);       // Wq^T  [8192][1024]
  size_t oWk  = alloc((size_t)1024 * 1024 * 2);       // Wk^T
  size_t oWv  = alloc((size_t)1024 * 1024 * 2);       // Wv^T
  size_t oWo  = alloc((size_t)1024 * 8192 * 2);       // Wo^T  [1024][8192]
  size_t oXb  = alloc((size_t)M * 1024 * 2);
  size_t oQ   = alloc((size_t)M * 8192 * 2);          // reused later as outflat
  size_t oK   = alloc((size_t)M * 1024 * 2);
  size_t oV   = alloc((size_t)M * 1024 * 2);
  size_t oQm  = alloc((size_t)B * Hh * N * 512 * 2);  // [z][query][512]
  size_t oKm  = alloc((size_t)B * N * 512 * 2);       // [b][key][512]   (Bt)
  size_t oVT  = alloc((size_t)B * 1024 * N * 2);      // [b][feat][key]  (Bt)
  size_t oS   = alloc((size_t)B * Hh * N * N * 2);    // scores / P
  size_t oAO  = alloc((size_t)B * Hh * N * 1024 * 2);

  unsigned short* Wq2  = (unsigned short*)(ws + oWq);
  unsigned short* Wk2  = (unsigned short*)(ws + oWk);
  unsigned short* Wv2  = (unsigned short*)(ws + oWv);
  unsigned short* Wo2  = (unsigned short*)(ws + oWo);
  unsigned short* xb   = (unsigned short*)(ws + oXb);
  unsigned short* qb   = (unsigned short*)(ws + oQ);
  unsigned short* kb   = (unsigned short*)(ws + oK);
  unsigned short* vb   = (unsigned short*)(ws + oV);
  unsigned short* qm   = (unsigned short*)(ws + oQm);
  unsigned short* km   = (unsigned short*)(ws + oKm);
  unsigned short* vT   = (unsigned short*)(ws + oVT);
  unsigned short* Sc   = (unsigned short*)(ws + oS);
  unsigned short* ao   = (unsigned short*)(ws + oAO);
  unsigned short* oflt = qb;                          // alias: q no longer needed

  dim3 blk(256);

  // 1) fuse weights (transposed, [Nout][Kin])
  {
    long t1 = (long)8192 * 1024;  // Wq: J=512, I=64
    fuse_weights_kernel<<<dim3((unsigned)((t1 + 255) / 256)), blk, 0, stream>>>(w_q, blade, Wq2, 512, 64, t1);
    long t2 = (long)1024 * 1024;
    fuse_weights_kernel<<<dim3((unsigned)((t2 + 255) / 256)), blk, 0, stream>>>(w_k, blade, Wk2, 64, 64, t2);
    fuse_weights_kernel<<<dim3((unsigned)((t2 + 255) / 256)), blk, 0, stream>>>(w_v, blade, Wv2, 64, 64, t2);
    long t3 = (long)1024 * 8192;  // Wo: J=64, I=512
    fuse_weights_kernel<<<dim3((unsigned)((t3 + 255) / 256)), blk, 0, stream>>>(w_o, blade, Wo2, 64, 512, t3);
  }

  // 2) x -> bf16
  {
    long n = M * 1024;
    cvt_f32_bf16_kernel<<<dim3((unsigned)((n + 255) / 256)), blk, 0, stream>>>(x, xb, n);
  }

  auto gemm = [&](const unsigned short* A, const unsigned short* Bt,
                  float* Cf, unsigned short* Cb,
                  int Mm, int Nn, int Kk,
                  long sA, int dA, long sB, int dB, long sC,
                  float alpha, int Z) {
    dim3 grid((unsigned)(Nn / 128), (unsigned)(Mm / 128), (unsigned)Z);
    gemm_bf16_wmma_kernel<<<grid, blk, 0, stream>>>(A, Bt, Cf, Cb, Mm, Nn, Kk,
                                                    sA, dA, sB, dB, sC, alpha);
  };

  // 3) projections
  gemm(xb, Wq2, nullptr, qb, (int)M, 8192, 1024, 0, 1, 0, 1, 0, 1.0f, 1);
  gemm(xb, Wk2, nullptr, kb, (int)M, 1024, 1024, 0, 1, 0, 1, 0, 1.0f, 1);
  gemm(xb, Wv2, nullptr, vb, (int)M, 1024, 1024, 0, 1, 0, 1, 0, 1.0f, 1);

  // 4) gathers for attention operands
  {
    long t = (long)B * Hh * N * 512;
    gather_qm_kernel<<<dim3((unsigned)((t + 255) / 256)), blk, 0, stream>>>(qb, qm, t);
    long t2 = (long)B * N * 512;
    gather_km_kernel<<<dim3((unsigned)((t2 + 255) / 256)), blk, 0, stream>>>(kb, km, t2);
    long t3 = (long)B * 1024 * N;
    transpose_v_kernel<<<dim3((unsigned)((t3 + 255) / 256)), blk, 0, stream>>>(vb, vT, t3);
  }

  // 5) scores = (1/sqrt(512)) * qm @ km^T   per z=(b*8+h), km shared across heads
  const float scale = 0.044194173824159216f;  // 1/sqrt(512)
  gemm(qm, km, nullptr, Sc, N, N, 512,
       (long)N * 512, 1, (long)N * 512, Hh, (long)N * N, scale, B * Hh);

  // 6) softmax rows (in place, fp32 math)
  softmax_rows_kernel<<<dim3((unsigned)(B * Hh * N)), blk, 0, stream>>>(Sc, N);

  // 7) attn_out = P @ V   per z, vT shared across heads
  gemm(Sc, vT, nullptr, ao, N, 1024, N,
       (long)N * N, 1, (long)1024 * N, Hh, (long)N * 1024, 1.0f, B * Hh);

  // 8) reorder heads-outer into flat [M, 8192] (reuses q buffer)
  {
    long t = (long)B * Hh * N * 1024;
    regather_out_kernel<<<dim3((unsigned)((t + 255) / 256)), blk, 0, stream>>>(ao, oflt, t);
  }

  // 9) output projection -> fp32 d_out
  gemm(oflt, Wo2, out, nullptr, (int)M, 1024, 8192, 0, 1, 0, 1, 0, 1.0f, 1);
}
